// SelfAttention_81398220194391
// MI455X (gfx1250) — matrix operand order repaired and checked by reading, not compiled, not run on hardware
//
#include <hip/hip_runtime.h>
#include <hip/hip_bf16.h>
#include <stdint.h>

// Problem constants (from reference)
#define E_DIM   1024
#define H_DIM   16
#define D_DIM   64
#define N_BATCH 4
#define L_SEQ   1024
#define NROWS   (N_BATCH * L_SEQ)   // 4096

typedef __attribute__((ext_vector_type(16))) __bf16    v16bf;
typedef __attribute__((ext_vector_type(2)))  __bf16    v2bf;
typedef __attribute__((ext_vector_type(8)))  float     v8f;
typedef __attribute__((ext_vector_type(8)))  uint32_t  v8u;
typedef __attribute__((ext_vector_type(4)))  int       v4i;

typedef __attribute__((address_space(1))) v4i  g_v4i;
typedef __attribute__((address_space(3))) v4i  l_v4i;

#if __has_builtin(__builtin_amdgcn_global_load_async_to_lds_b128)
#define HAVE_ASYNC_LDS 1
#else
#define HAVE_ASYNC_LDS 0
#endif

// ---------------------------------------------------------------------------
// fp32 -> bf16 (round to nearest even), raw u16 bits
static __device__ __forceinline__ uint16_t f2bf(float f) {
    uint32_t u = __float_as_uint(f);
    u += 0x7FFFu + ((u >> 16) & 1u);
    return (uint16_t)(u >> 16);
}

// pack two fp32 -> one dword of two bf16 (uses v_cvt_pk_bf16_f32 when present)
static __device__ __forceinline__ uint32_t pack_bf16(float lo, float hi) {
#if __has_builtin(__builtin_amdgcn_cvt_pk_bf16_f32)
    v2bf r = __builtin_amdgcn_cvt_pk_bf16_f32(lo, hi);
    return __builtin_bit_cast(uint32_t, r);
#else
    return (uint32_t)f2bf(lo) | ((uint32_t)f2bf(hi) << 16);
#endif
}

static __device__ __forceinline__ v8f wmma_bf16(v8u a, v8u b, v8f c) {
    return __builtin_amdgcn_wmma_f32_16x16x32_bf16(
        false, __builtin_bit_cast(v16bf, a),
        false, __builtin_bit_cast(v16bf, b),
        (short)0, c, false, false);
}

// A fragment (16x32 bf16, M x K): element (m,k) at base[m*rowStride + k].
// Layout (ISA 7.12.2): lane m = lane&15; kb = (lane>=16)*8;
// vgpr j<4 -> K = kb+2j..; vgpr j>=4 -> K = kb+16+2(j-4)..
static __device__ __forceinline__ v8u load_a_frag(const uint16_t* base, int rowStride, int lane) {
    const int row = lane & 15;
    const int kb  = (lane >> 4) << 3;
    const uint16_t* p = base + row * rowStride + kb;
    v8u r;
#pragma unroll
    for (int j = 0; j < 8; ++j) {
        int k = (j < 4) ? (2 * j) : (16 + 2 * (j - 4));
        r[j] = *(const uint32_t*)(p + k);
    }
    return r;
}

// A fragment from fp32 source (converted on the fly)
static __device__ __forceinline__ v8u load_a_frag_f32(const float* base, int rowStride, int lane) {
    const int row = lane & 15;
    const int kb  = (lane >> 4) << 3;
    const float* p = base + row * rowStride + kb;
    v8u r;
#pragma unroll
    for (int j = 0; j < 8; ++j) {
        int k = (j < 4) ? (2 * j) : (16 + 2 * (j - 4));
        r[j] = pack_bf16(p[k], p[k + 1]);
    }
    return r;
}

// B fragment (32x16 bf16, K x N): element (k,n) at base[n*colStride + k].
// Layout: lane n = lane&15; kb = (lane>=16)*8; vgpr j<4 -> K = kb+2j..; vgpr j>=4 -> K = kb+16+2(j-4)..
static __device__ __forceinline__ v8u load_b_frag(const uint16_t* base, int colStride, int lane) {
    const int col = lane & 15;
    const int kb  = (lane >> 4) << 3;
    const uint16_t* p = base + col * colStride + kb;
    v8u r;
#pragma unroll
    for (int j = 0; j < 8; ++j) r[j] = *(const uint32_t*)(p + ((j < 4) ? (2 * j) : (16 + 2 * (j - 4))));
    return r;
}

// async 16-byte global -> LDS copy (ASYNCcnt-tracked) with sync fallback
static __device__ __forceinline__ void copy16_g2lds(const uint16_t* g, uint16_t* lds) {
#if HAVE_ASYNC_LDS
    __builtin_amdgcn_global_load_async_to_lds_b128(
        (g_v4i*)g, (l_v4i*)lds, 0, 0);
#else
    *(uint4*)lds = *(const uint4*)g;
#endif
}

static __device__ __forceinline__ void wait_async_lds() {
#if HAVE_ASYNC_LDS
#if __has_builtin(__builtin_amdgcn_s_wait_asynccnt)
    __builtin_amdgcn_s_wait_asynccnt(0);
#else
    asm volatile("s_wait_asynccnt 0" ::: "memory");
#endif
#endif
}

// ---------------------------------------------------------------------------
// fp32 -> bf16 elementwise convert (weights)
__global__ void cvt_f32_bf16_kernel(const float* __restrict__ src,
                                    uint16_t* __restrict__ dst, int n) {
    int i = blockIdx.x * blockDim.x + threadIdx.x;
    if (i < n) dst[i] = f2bf(src[i]);
}

// ---------------------------------------------------------------------------
// Per-head projection: out[l,o] = sum_d in[l, h*64+d] * W[o,d]
// 4 waves/block; wave w computes the 16-row x 16-col tile at oTile=w.
// transposed==0: out bf16 [N,H,L,64]; transposed==1: out bf16 [N,H,64,L]
__global__ void __launch_bounds__(128)
proj_kernel(const float* __restrict__ in,     // [NROWS, E]
            const uint16_t* __restrict__ Wb,  // [64, 64] bf16, row-major [o][d]
            uint16_t* __restrict__ out,
            int transposed) {
    const int lane = threadIdx.x & 31;
    const int wv   = threadIdx.x >> 5;        // output col tile 0..3
    const int rowTile = blockIdx.x;           // 0..255
    const int h = blockIdx.y;

    const float* inBase = in + (size_t)(rowTile * 16) * E_DIM + h * D_DIM;

    v8f acc = {};
#pragma unroll
    for (int s = 0; s < 2; ++s) {            // K = 64 in two 32-steps
        v8u a = load_a_frag_f32(inBase + s * 32, E_DIM, lane);
        v8u b = load_b_frag(Wb + wv * 16 * D_DIM + s * 32, D_DIM, lane);
        acc = wmma_bf16(a, b, acc);
    }

    const int col = wv * 16 + (lane & 15);
#pragma unroll
    for (int v = 0; v < 8; ++v) {
        int m = v + ((lane >> 4) << 3);
        int gRow = rowTile * 16 + m;
        int n = gRow >> 10;                   // / L_SEQ
        int l = gRow & (L_SEQ - 1);
        uint16_t val = f2bf(acc[v]);
        if (!transposed)
            out[((size_t)(n * H_DIM + h) * L_SEQ + l) * D_DIM + col] = val;
        else
            out[((size_t)(n * H_DIM + h) * D_DIM + col) * L_SEQ + l] = val;
    }
}

// ---------------------------------------------------------------------------
// Flash attention: 4 waves/block share one (n,h); wave wv owns q-tile
// blockIdx.x*4+wv. Each 32-key block's K tile (32x64, contiguous 4 KB) and
// V tile (64x32) are async-staged into LDS once per block and shared by all
// 4 waves. Online softmax; P transposed C-layout -> A-layout through LDS.
__global__ void __launch_bounds__(128)
attn_kernel(const uint16_t* __restrict__ Qp,   // [N,H,L,64]
            const uint16_t* __restrict__ Kp,   // [N,H,L,64]
            const uint16_t* __restrict__ Vp,   // [N,H,64,L]
            const int* __restrict__ mask,      // [N, L]
            uint16_t* __restrict__ AO) {       // [N*L, E] bf16
    __shared__ __align__(16) uint16_t Klds[32][64];   // [key][d]  4 KB
    __shared__ __align__(16) uint16_t Vlds[64][32];   // [d][key]  4 KB
    __shared__ __align__(16) uint16_t Plds[4][16][32];// per-wave P 4 KB

    const int tid  = threadIdx.x;
    const int lane = tid & 31;
    const int wv   = tid >> 5;
    const int h = blockIdx.y, n = blockIdx.z;
    const int qTile = blockIdx.x * 4 + wv;

    const size_t nh = (size_t)(n * H_DIM + h);
    const uint16_t* Qb = Qp + (nh * L_SEQ + qTile * 16) * D_DIM;
    const uint16_t* Kb = Kp + nh * L_SEQ * D_DIM;
    const uint16_t* Vb = Vp + nh * (size_t)D_DIM * L_SEQ;
    const int* mrow = mask + n * L_SEQ;

    v8u aq0 = load_a_frag(Qb, D_DIM, lane);
    v8u aq1 = load_a_frag(Qb + 32, D_DIM, lane);

    v8f o0 = {}, o1 = {}, o2 = {}, o3 = {};
    float mrun[8], lrun[8];
#pragma unroll
    for (int v = 0; v < 8; ++v) { mrun[v] = -3.0e38f; lrun[v] = 0.0f; }

    const float scale = 0.03125f;             // 1/sqrt(E) = 1/32
    const int col = lane & 15;

    for (int kb = 0; kb < L_SEQ; kb += 32) {
        // ---- async stage K tile (contiguous) and V tile into LDS ----
        const uint16_t* Ksrc = Kb + (size_t)kb * D_DIM;   // 2048 elems, contiguous
#pragma unroll
        for (int i = 0; i < 2; ++i) {
            int chunk = tid + i * 128;                    // 0..255 x 16B
            copy16_g2lds(Ksrc + chunk * 8, &Klds[0][0] + chunk * 8);
            int vrow = chunk >> 2;                        // d index 0..63
            int voff = (chunk & 3) * 8;                   // key offset
            copy16_g2lds(Vb + (size_t)vrow * L_SEQ + kb + voff, &Vlds[vrow][voff]);
        }
        wait_async_lds();
        __syncthreads();

        // ---- S = Q . K^T (two 16-key tiles, K-dim 64 in 2 steps) ----
        v8f s0 = {}, s1 = {};
        s0 = wmma_bf16(aq0, load_b_frag(&Klds[0][0],       64, lane), s0);
        s0 = wmma_bf16(aq1, load_b_frag(&Klds[0][0]  + 32, 64, lane), s0);
        s1 = wmma_bf16(aq0, load_b_frag(&Klds[16][0],      64, lane), s1);
        s1 = wmma_bf16(aq1, load_b_frag(&Klds[16][0] + 32, 64, lane), s1);

        // ---- key mask (column-wise) + scale ----
        const bool z0 = (mrow[kb + col] == 0);
        const bool z1 = (mrow[kb + 16 + col] == 0);
        float e0[8], e1[8];
#pragma unroll
        for (int v = 0; v < 8; ++v) {
            e0[v] = (z0 ? -1.0e20f : s0[v]) * scale;
            e1[v] = (z1 ? -1.0e20f : s1[v]) * scale;
        }

        // ---- online softmax: row stats share the C-fragment layout ----
        float alpha[8];
#pragma unroll
        for (int v = 0; v < 8; ++v) {
            float bm = fmaxf(e0[v], e1[v]);
#pragma unroll
            for (int off = 1; off < 16; off <<= 1)
                bm = fmaxf(bm, __shfl_xor(bm, off, 16));
            float mnew = fmaxf(mrun[v], bm);
            alpha[v] = __expf(mrun[v] - mnew);
            mrun[v] = mnew;
            e0[v] = __expf(e0[v] - mnew);
            e1[v] = __expf(e1[v] - mnew);
            float rs = e0[v] + e1[v];
#pragma unroll
            for (int off = 1; off < 16; off <<= 1)
                rs += __shfl_xor(rs, off, 16);
            lrun[v] = lrun[v] * alpha[v] + rs;
        }

        // ---- rescale O accumulators, stage P (C-layout) into LDS ----
#pragma unroll
        for (int v = 0; v < 8; ++v) {
            o0[v] *= alpha[v]; o1[v] *= alpha[v];
            o2[v] *= alpha[v]; o3[v] *= alpha[v];
            int m = v + ((lane >> 4) << 3);
            uint32_t pk = pack_bf16(e0[v], e1[v]);
            Plds[wv][m][col]      = (uint16_t)(pk & 0xFFFFu);
            Plds[wv][m][16 + col] = (uint16_t)(pk >> 16);
        }
        __syncthreads();                       // order P store vs punned load
        v8u pa = load_a_frag(&Plds[wv][0][0], 32, lane);

        // ---- O += P . V (V fragment from LDS, keys contiguous) ----
        o0 = wmma_bf16(pa, load_b_frag(&Vlds[ 0][0], 32, lane), o0);
        o1 = wmma_bf16(pa, load_b_frag(&Vlds[16][0], 32, lane), o1);
        o2 = wmma_bf16(pa, load_b_frag(&Vlds[32][0], 32, lane), o2);
        o3 = wmma_bf16(pa, load_b_frag(&Vlds[48][0], 32, lane), o3);

        __syncthreads();                       // protect K/V tiles for next iter
    }

    // normalize and write bf16 back to [N,L,E] interleaved by head
#pragma unroll
    for (int v = 0; v < 8; ++v) {
        float inv = 1.0f / lrun[v];
        int m = v + ((lane >> 4) << 3);
        int q = qTile * 16 + m;
        size_t base = ((size_t)(n * L_SEQ + q)) * E_DIM + h * D_DIM;
        AO[base +  0 + col] = f2bf(o0[v] * inv);
        AO[base + 16 + col] = f2bf(o1[v] * inv);
        AO[base + 32 + col] = f2bf(o2[v] * inv);
        AO[base + 48 + col] = f2bf(o3[v] * inv);
    }
}

// ---------------------------------------------------------------------------
// Output projection: out[r,c] = sum_e AO[r,e] * Wo[c,e] + bo[c], fp32 out.
// Block = 128 threads (4 waves) computes a 64x64 tile. A and Wo tiles are
// staged through LDS with async global->LDS copies (ASYNCcnt) so global
// traffic is coalesced and the Wo tile is shared by all 4 waves.
__global__ void __launch_bounds__(128)
out_gemm_kernel(const uint16_t* __restrict__ AO,   // [4096, 1024] bf16
                const uint16_t* __restrict__ Wob,  // [1024, 1024] bf16 [c][e]
                const float* __restrict__ bo,      // [1024]
                float* __restrict__ out) {         // [4096, 1024] f32
    __shared__ __align__(16) uint16_t Alds[64][64];   // [row][e]  8 KB
    __shared__ __align__(16) uint16_t Blds[64][64];   // [c][e]    8 KB

    const int tid  = threadIdx.x;
    const int lane = tid & 31;
    const int wv   = tid >> 5;
    const int r0 = blockIdx.x * 64;
    const int c0 = blockIdx.y * 64;

    v8f acc[4] = {{}, {}, {}, {}};

    for (int ks = 0; ks < E_DIM; ks += 64) {
        // stage 64x64 A tile and 64x64 Wo tile: 512 x 16B chunks each,
        // 4 chunks per thread per tile, fully coalesced.
#pragma unroll
        for (int i = 0; i < 4; ++i) {
            int chunk = tid + i * 128;
            int row = chunk >> 3;
            int colE = (chunk & 7) * 8;
            copy16_g2lds(AO  + (size_t)(r0 + row) * E_DIM + ks + colE, &Alds[row][colE]);
            copy16_g2lds(Wob + (size_t)(c0 + row) * E_DIM + ks + colE, &Blds[row][colE]);
        }
        wait_async_lds();
        __syncthreads();

        // wave wv owns output rows r0 + wv*16 .. +15
        const uint16_t* Arow = &Alds[wv * 16][0];
#pragma unroll
        for (int kk = 0; kk < 64; kk += 32) {
            v8u a = load_a_frag(Arow + kk, 64, lane);
#pragma unroll
            for (int t = 0; t < 4; ++t) {
                v8u b = load_b_frag(&Blds[t * 16][0] + kk, 64, lane);
                acc[t] = wmma_bf16(a, b, acc[t]);
            }
        }
        __syncthreads();
    }

    const int colN = lane & 15;
#pragma unroll
    for (int t = 0; t < 4; ++t) {
        int c = c0 + t * 16 + colN;
        float bias = bo[c];
#pragma unroll
        for (int v = 0; v < 8; ++v) {
            int m = v + ((lane >> 4) << 3);
            out[(size_t)(r0 + wv * 16 + m) * E_DIM + c] = acc[t][v] + bias;
        }
    }
}

// ---------------------------------------------------------------------------
extern "C" void kernel_launch(void* const* d_in, const int* in_sizes, int n_in,
                              void* d_out, int out_size, void* d_ws, size_t ws_size,
                              hipStream_t stream) {
    const float* values  = (const float*)d_in[0];
    const float* keys    = (const float*)d_in[1];
    const float* queries = (const float*)d_in[2];
    const int*   mask    = (const int*)d_in[3];
    const float* Wv      = (const float*)d_in[4];
    const float* Wk      = (const float*)d_in[5];
    const float* Wq      = (const float*)d_in[6];
    const float* Wo      = (const float*)d_in[7];
    const float* bo      = (const float*)d_in[8];
    float* out = (float*)d_out;

    // Workspace layout (bf16 elems): Qp, Kp, Vp, AO (4M each), Wq/Wk/Wv (4K), Wo (1M)
    uint16_t* ws = (uint16_t*)d_ws;
    const size_t PROJ = (size_t)N_BATCH * H_DIM * L_SEQ * D_DIM;  // 4 Mi elems
    uint16_t* Qp  = ws;
    uint16_t* Kp  = Qp  + PROJ;
    uint16_t* Vp  = Kp  + PROJ;
    uint16_t* AO  = Vp  + PROJ;
    uint16_t* Wqb = AO  + PROJ;
    uint16_t* Wkb = Wqb + D_DIM * D_DIM;
    uint16_t* Wvb = Wkb + D_DIM * D_DIM;
    uint16_t* Wob = Wvb + D_DIM * D_DIM;

    // 1) weights -> bf16
    cvt_f32_bf16_kernel<<<(D_DIM * D_DIM + 255) / 256, 256, 0, stream>>>(Wq, Wqb, D_DIM * D_DIM);
    cvt_f32_bf16_kernel<<<(D_DIM * D_DIM + 255) / 256, 256, 0, stream>>>(Wk, Wkb, D_DIM * D_DIM);
    cvt_f32_bf16_kernel<<<(D_DIM * D_DIM + 255) / 256, 256, 0, stream>>>(Wv, Wvb, D_DIM * D_DIM);
    cvt_f32_bf16_kernel<<<(E_DIM * E_DIM + 255) / 256, 256, 0, stream>>>(Wo, Wob, E_DIM * E_DIM);

    // 2) Q/K/V projections (V stored transposed [N,H,64,L])
    dim3 gproj(NROWS / 16, H_DIM);
    proj_kernel<<<gproj, 128, 0, stream>>>(queries, Wqb, Qp, 0);
    proj_kernel<<<gproj, 128, 0, stream>>>(keys,    Wkb, Kp, 0);
    proj_kernel<<<gproj, 128, 0, stream>>>(values,  Wvb, Vp, 1);

    // 3) flash attention (async-LDS staged K/V tiles)
    dim3 gattn(L_SEQ / 64, H_DIM, N_BATCH);
    attn_kernel<<<gattn, 128, 0, stream>>>(Qp, Kp, Vp, mask, AO);

    // 4) output projection + bias (async-LDS staged GEMM)
    dim3 gout(NROWS / 64, E_DIM / 64);
    out_gemm_kernel<<<gout, 128, 0, stream>>>(AO, Wob, bo, out);
}
